// Model_14482629722722
// MI455X (gfx1250) — matrix operand order repaired and checked by reading, not compiled, not run on hardware
//
#include <hip/hip_runtime.h>
#include <hip/hip_bf16.h>

// Problem constants (fixed by setup_inputs)
#define BB 16
#define KK 32
#define LL 1024
#define DD 128
#define NL 48          // new_l
#define NS 49          // NL + 1 (invalid bucket)

// Output layout (flat concat, floats):
//   new_h0        : BB*KK*NL*DD = 3,145,728   @ 0
//   new_event_time: BB*KK*NL    =    24,576   @ 3,145,728
//   new_pad_mask  : BB*KK*NL    =    24,576   @ 3,170,304
//   almat         : BB*KK*LL*NL = 25,165,824  @ 3,194,880
#define NH_OFF  0
#define NET_OFF 3145728
#define PAD_OFF 3170304
#define ALM_OFF 3194880

typedef _Float16 v16h __attribute__((ext_vector_type(16)));
typedef float    v8f  __attribute__((ext_vector_type(8)));

// ---------------------------------------------------------------------------
// Kernel A: per batch (one 1024-thread block) compute seg[], CSR (offsets +
// index list), counts, tsum (deterministic serial order), and write
// new_event_time. LDS-resident, trivial cost.
// ---------------------------------------------------------------------------
__global__ __launch_bounds__(LL) void seg_build_kernel(
    const float* __restrict__ et, int* __restrict__ seg,
    int* __restrict__ idx, int* __restrict__ offs,
    float* __restrict__ cnts, float* __restrict__ tsums,
    float* __restrict__ net_out) {
  __shared__ float sev[LL];
  __shared__ int   ssg[LL];
  __shared__ int   cnt[NL];
  __shared__ float ts[NL];
  __shared__ int   off[NS];
  __shared__ int   cur[NL];

  const int b = blockIdx.x;
  const int l = threadIdx.x;

  float t = et[b * LL + l];
  bool valid = (t >= 0.0f) && (t < (float)NL);
  int s = valid ? (int)floorf(t) : NL;
  if (s < 0 || s > NL) s = NL;
  sev[l] = t;
  ssg[l] = s;
  seg[b * LL + l] = s;
  if (l < NL) cur[l] = 0;
  __syncthreads();

  // Deterministic per-segment count + time-sum (fixed ascending-l order).
  if (l < NL) {
    int c = 0;
    float sum = 0.0f;
    for (int i = 0; i < LL; ++i) {
      if (ssg[i] == l) { c++; sum += sev[i]; }
    }
    cnt[l] = c;
    ts[l] = sum;
  }
  __syncthreads();

  if (l == 0) {
    int acc = 0;
    for (int i = 0; i < NL; ++i) { off[i] = acc; acc += cnt[i]; }
    off[NL] = acc;
  }
  __syncthreads();

  // CSR placement (order within a segment is arbitrary; consumer is max()).
  if (s < NL) {
    int pos = off[s] + atomicAdd(&cur[s], 1);
    idx[b * LL + pos] = l;
  }

  if (l < NL) {
    cnts[b * NL + l]  = (float)cnt[l];
    tsums[b * NL + l] = ts[l];
    offs[b * NS + l]  = off[l];
  }
  if (l == 0) offs[b * NS + NL] = off[NL];
  __syncthreads();

  // new_event_time[b][k][j] = tsum[j]/max(cnt[j],1), broadcast over k.
  for (int t2 = l; t2 < KK * NL; t2 += blockDim.x) {
    int k = t2 / NL, j = t2 % NL;
    float denom = fmaxf((float)cnt[j], 1.0f);
    net_out[(b * KK + k) * NL + j] = ts[j] / denom;
  }
}

// ---------------------------------------------------------------------------
// Kernel B: segment max.  One block per (b,k,s); 128 threads = one D row.
// CSR gather reads each h0 row exactly once (256 MiB total), fully coalesced
// (512 B contiguous per l).  global_prefetch_b8 hides gather latency.
// ---------------------------------------------------------------------------
__global__ __launch_bounds__(DD) void seg_max_kernel(
    const float* __restrict__ h0, const int* __restrict__ idx,
    const int* __restrict__ offs, float* __restrict__ new_h0) {
  __shared__ int lidx[LL];
  const int bid = blockIdx.x;
  const int s = bid % NL;
  const int k = (bid / NL) % KK;
  const int b = bid / (NL * KK);

  const int o0 = offs[b * NS + s];
  const int o1 = offs[b * NS + s + 1];
  const int count = o1 - o0;

  for (int e = threadIdx.x; e < count; e += blockDim.x)
    lidx[e] = idx[b * LL + o0 + e];
  __syncthreads();

  const int d = threadIdx.x;
  const float* base = h0 + (size_t)(b * KK + k) * LL * DD + d;

  float m = -__builtin_inff();
  for (int e = 0; e < count; ++e) {
    if (e + 1 < count)
      __builtin_prefetch(base + (size_t)lidx[e + 1] * DD, 0, 0);
    m = fmaxf(m, base[(size_t)lidx[e] * DD]);
  }
  if (count < LL) m = fmaxf(m, 0.0f);  // include_zero relu (empty seg -> 0)

  new_h0[((size_t)(b * KK + k) * NL + s) * DD + d] = m;
}

// ---------------------------------------------------------------------------
// Kernel C: almat one-hot writes, float4-vectorized, fully coalesced (96 MiB).
// ---------------------------------------------------------------------------
__global__ __launch_bounds__(256) void almat_kernel(
    const int* __restrict__ seg, float4* __restrict__ alm4) {
  const int tid = blockIdx.x * blockDim.x + threadIdx.x;  // BB*KK*LL*12 threads
  const int j4  = tid % 12;
  const int l   = (tid / 12) % LL;
  const int rem = tid / (12 * LL);
  const int b   = rem / KK;

  const int s = seg[b * LL + l];
  const int j0 = j4 * 4;
  float4 v;
  v.x = (s == j0 + 0) ? 1.0f : 0.0f;
  v.y = (s == j0 + 1) ? 1.0f : 0.0f;
  v.z = (s == j0 + 2) ? 1.0f : 0.0f;
  v.w = (s == j0 + 3) ? 1.0f : 0.0f;
  alm4[tid] = v;
}

// ---------------------------------------------------------------------------
// Kernel D: new_pad_mask via WMMA.  P = npm (32x1024) x almat (1024x48),
// P[k][j] = msum[j][k].  Inputs are exactly 0/1 -> f16 operands + f32
// accumulation are bit-exact.  One wave per batch; B operand synthesized
// in-register from seg cached in LDS.
// Layouts per ISA 7.12.2 (16-bit A 16x32; B as halved 64x16 pattern; f32 D).
// ---------------------------------------------------------------------------
__global__ __launch_bounds__(32) void pad_wmma_kernel(
    const float* __restrict__ npm, const int* __restrict__ seg,
    const float* __restrict__ cnts, float* __restrict__ pad_out) {
  __shared__ int   sseg[LL];
  __shared__ float sden[NL];

  const int b = blockIdx.x;
  const int lane = threadIdx.x;

  for (int i = lane; i < LL; i += 32) sseg[i] = seg[b * LL + i];
  for (int i = lane; i < NL; i += 32) sden[i] = fmaxf(cnts[b * NL + i], 1.0f);
  __syncthreads();

  const int n = lane & 15;      // A: row m (0..15); B: col n; D: col n
  const int half = lane >> 4;   // lane-half selects K interleave / M block

  v8f acc[2][3];
#pragma unroll
  for (int tm = 0; tm < 2; ++tm)
#pragma unroll
    for (int tn = 0; tn < 3; ++tn)
#pragma unroll
      for (int e = 0; e < 8; ++e) acc[tm][tn][e] = 0.0f;

  for (int kc = 0; kc < LL; kc += 32) {
    // A tiles: elements 0..7 <- K = kc+half*8+i ; 8..15 <- K = kc+16+half*8+i
    v16h aT[2];
#pragma unroll
    for (int tm = 0; tm < 2; ++tm) {
      const int row = tm * 16 + n;  // npm row (k index)
      const float* rp = npm + (size_t)(b * KK + row) * LL;
      const float4* g0 = (const float4*)(rp + kc + half * 8);
      const float4* g1 = (const float4*)(rp + kc + 16 + half * 8);
      float4 p0 = g0[0], p1 = g0[1], q0 = g1[0], q1 = g1[1];
      aT[tm][0] = (_Float16)p0.x;  aT[tm][1] = (_Float16)p0.y;
      aT[tm][2] = (_Float16)p0.z;  aT[tm][3] = (_Float16)p0.w;
      aT[tm][4] = (_Float16)p1.x;  aT[tm][5] = (_Float16)p1.y;
      aT[tm][6] = (_Float16)p1.z;  aT[tm][7] = (_Float16)p1.w;
      aT[tm][8]  = (_Float16)q0.x; aT[tm][9]  = (_Float16)q0.y;
      aT[tm][10] = (_Float16)q0.z; aT[tm][11] = (_Float16)q0.w;
      aT[tm][12] = (_Float16)q1.x; aT[tm][13] = (_Float16)q1.y;
      aT[tm][14] = (_Float16)q1.z; aT[tm][15] = (_Float16)q1.w;
    }
    // B tiles: element i <- K = kc+half*8+i (i<8), kc+16+half*8+(i-8) (i>=8) ; value = (seg[K]==col)
    v16h bT[3];
#pragma unroll
    for (int tn = 0; tn < 3; ++tn) {
      const int col = tn * 16 + n;
#pragma unroll
      for (int i = 0; i < 16; ++i) {
        const int kk = kc + (i < 8 ? half * 8 + i : 16 + half * 8 + (i - 8));
        bT[tn][i] = (sseg[kk] == col) ? (_Float16)1.0f : (_Float16)0.0f;
      }
    }
#pragma unroll
    for (int tm = 0; tm < 2; ++tm)
#pragma unroll
      for (int tn = 0; tn < 3; ++tn)
        acc[tm][tn] = __builtin_amdgcn_wmma_f32_16x16x32_f16(
            false, aT[tm], false, bT[tn], (short)0, acc[tm][tn], false, false);
  }

  // D layout: element r holds M = half*8 + r, N = n.
#pragma unroll
  for (int tm = 0; tm < 2; ++tm)
#pragma unroll
    for (int tn = 0; tn < 3; ++tn)
#pragma unroll
      for (int r = 0; r < 8; ++r) {
        const int k = tm * 16 + half * 8 + r;
        const int j = tn * 16 + n;
        pad_out[(size_t)(b * KK + k) * NL + j] = acc[tm][tn][r] / sden[j];
      }
}

// ---------------------------------------------------------------------------
extern "C" void kernel_launch(void* const* d_in, const int* in_sizes, int n_in,
                              void* d_out, int out_size, void* d_ws, size_t ws_size,
                              hipStream_t stream) {
  const float* h0  = (const float*)d_in[0];  // (16,32,1024,128) f32
  const float* et  = (const float*)d_in[1];  // (16,1024) f32
  const float* npm = (const float*)d_in[2];  // (16,32,1024) f32
  // d_in[3] = new_l (==48, baked into constants)

  float* out = (float*)d_out;

  // Workspace layout (~140 KB)
  int*   seg   = (int*)d_ws;           // BB*LL
  int*   idx   = seg + BB * LL;        // BB*LL
  int*   offs  = idx + BB * LL;        // BB*NS
  float* cnts  = (float*)(offs + BB * NS);  // BB*NL
  float* tsums = cnts + BB * NL;            // BB*NL

  seg_build_kernel<<<BB, LL, 0, stream>>>(et, seg, idx, offs, cnts, tsums,
                                          out + NET_OFF);

  seg_max_kernel<<<BB * KK * NL, DD, 0, stream>>>(h0, idx, offs, out + NH_OFF);

  almat_kernel<<<(BB * KK * LL * 12) / 256, 256, 0, stream>>>(
      seg, (float4*)(out + ALM_OFF));

  pad_wmma_kernel<<<BB, 32, 0, stream>>>(npm, seg, cnts, out + PAD_OFF);
}
